// MMDFeatureFusion_13924283973744
// MI455X (gfx1250) — matrix operand order repaired and checked by reading, not compiled, not run on hardware
//
#include <hip/hip_runtime.h>
#include <math.h>

typedef _Float16 f16;
typedef __attribute__((ext_vector_type(16))) _Float16 v16h;
typedef __attribute__((ext_vector_type(8)))  _Float16 v8h;
typedef __attribute__((ext_vector_type(8)))  float    v8f;

#define BB 4
#define CCH 256
#define HH 28
#define WWD 28
#define NSP 784            // H*W
#define NG 8
#define CG 32
#define NHEADS 8
#define HCH 32
#define NBH 32             // B*HEADS == B*G
#define KP 800             // padded K for attn@v (784 -> 800)

// ---------------------------------------------------------------- helpers
__device__ __forceinline__ float gelu_f(float x) {
  return 0.5f * x * (1.f + erff(x * 0.70710678118654752f));
}

__device__ __forceinline__ float bilin2d(const float* img, int Wd, int Hd,
                                         float gx, float gy) {
  float x0f = floorf(gx), y0f = floorf(gy);
  int   x0  = (int)x0f,   y0  = (int)y0f;
  float wx1 = gx - x0f, wy1 = gy - y0f;
  float wx0 = 1.f - wx1, wy0 = 1.f - wy1;
  float acc = 0.f;
#pragma unroll
  for (int j = 0; j < 2; ++j) {
#pragma unroll
    for (int i = 0; i < 2; ++i) {
      int xi = x0 + i, yi = y0 + j;
      if (xi >= 0 && xi < Wd && yi >= 0 && yi < Hd)
        acc += img[yi * Wd + xi] * ((i ? wx1 : wx0) * (j ? wy1 : wy0));
    }
  }
  return acc;
}

union H16 { v16h v; v8h h[2]; };

// ---------------------------------------------------------------- WMMA GEMM
// D[z,m,n] = scale * sum_k A[z,m,k]*B[z,k,n] (+bias[m]) (+resid)
// REQUIREMENTS: A k-stride == 1, B k-stride == 1, K%32==0, (M/16)%MBLK==0.
// MBLK m-subtiles per wave, straight-line (no divergent guards at all).
// Batch index z decomposes as z = zq*zdiv + zr with independent strides.
template <int MBLK>
__global__ void wmma_gemm_f16(
    const f16* __restrict__ A,  long ars, long abs1, long abs2,
    const f16* __restrict__ Bp, long bcs, long bbs1, long bbs2,
    float* __restrict__ Cf, long crs, long ccs, long cbs1, long cbs2,
    f16* __restrict__ Ch, long hrs, long hcs, long hbs1, long hbs2,
    const float* __restrict__ bias,
    const float* __restrict__ resid, long rbs1, long rbs2,
    float scale, int K, int mtb, int nt, int zdiv, int batches) {
  int wid  = blockIdx.x * (blockDim.x >> 5) + (threadIdx.x >> 5);
  int lane = threadIdx.x & 31;
  int tpb  = mtb * nt;
  int z    = wid / tpb;
  if (z >= batches) return;
  int t  = wid - z * tpb;
  int tm = t / nt, tn = t - tm * nt;
  int zq = z / zdiv, zr = z - zq * zdiv;
  int lrow = lane & 15, lhi = lane >> 4;
  int n0 = tn << 4, mbase = tm * (MBLK << 4);

  const f16* Ab = A + (long)zq * abs1 + (long)zr * abs2;
  const f16* Bq = Bp + (long)zq * bbs1 + (long)zr * bbs2 +
                  (long)(n0 + lrow) * bcs + (lhi << 3);
  const f16* Ai[MBLK];
#pragma unroll
  for (int i = 0; i < MBLK; ++i)
    Ai[i] = Ab + (long)(mbase + (i << 4) + lrow) * ars + (lhi << 3);

  v8f acc[MBLK];
#pragma unroll
  for (int i = 0; i < MBLK; ++i) acc[i] = (v8f){};

  int ksteps = K >> 5;
  for (int ks = 0; ks < ksteps; ++ks) {
    int k0 = ks << 5;
    // B fragment: lane holds N=lane%16, e<8 -> k0+(lane/16)*8+e, e>=8 -> k0+16+(lane/16)*8+(e-8)
    H16 bu;
    bu.h[0] = *(const v8h*)(Bq + k0);
    bu.h[1] = *(const v8h*)(Bq + k0 + 16);
#pragma unroll
    for (int i = 0; i < MBLK; ++i) {
      // A fragment: lane holds M=lane%16, e<8 -> k0+(lane/16)*8+e,
      //                                  e>=8 -> k0+16+(lane/16)*8+(e-8)
      H16 au;
      au.h[0] = *(const v8h*)(Ai[i] + k0);
      au.h[1] = *(const v8h*)(Ai[i] + k0 + 16);
      acc[i] = __builtin_amdgcn_wmma_f32_16x16x32_f16(
          false, au.v, false, bu.v, (short)0, acc[i], false, false);
    }
  }
  // C/D: VGPR r -> M = r + (lane/16)*8 ; N = lane%16
  int n = n0 + lrow;
#pragma unroll
  for (int i = 0; i < MBLK; ++i) {
#pragma unroll
    for (int r = 0; r < 8; ++r) {
      int m = mbase + (i << 4) + r + (lhi << 3);
      float v = acc[i][r] * scale;
      if (bias)  v += bias[m];
      if (resid) v += resid[(long)zq * rbs1 + (long)zr * rbs2 +
                            (long)m * crs + (long)n * ccs];
      if (Cf) Cf[(long)zq * cbs1 + (long)zr * cbs2 +
                 (long)m * crs + (long)n * ccs] = v;
      if (Ch) Ch[(long)zq * hbs1 + (long)zr * hbs2 +
                 (long)m * hrs + (long)n * hcs] = (f16)v;
    }
  }
}

// ---------------------------------------------------------------- small kernels
__global__ void cvt_f32_to_f16(const float* __restrict__ in,
                               f16* __restrict__ out, long n) {
  long i = (long)blockIdx.x * blockDim.x + threadIdx.x;
  if (i < n) out[i] = (f16)in[i];
}

__global__ void zero_pad_v(f16* __restrict__ vpad) {
  int i = blockIdx.x * blockDim.x + threadIdx.x;     // BB*CCH*16
  if (i >= BB * CCH * 16) return;
  int j = i & 15, c = (i >> 4) & 255, b = i >> 12;
  vpad[(long)b * (CCH * KP) + (long)c * KP + NSP + j] = (f16)0;
}

// avg & max over spatial for concat(rgb,h): y0[b][0:512]=avg, [512:1024]=max
__global__ void cr_pool(const float* __restrict__ rgb,
                        const float* __restrict__ h,
                        float* __restrict__ y0) {
  int b  = blockIdx.x >> 9;
  int c2 = blockIdx.x & 511;
  const float* src = (c2 < CCH) ? rgb + ((long)b * CCH + c2) * NSP
                                : h   + ((long)b * CCH + (c2 - CCH)) * NSP;
  int t = threadIdx.x;
  float s = 0.f, mx = -3.402823466e38f;
  for (int i = t; i < NSP; i += blockDim.x) {
    float v = src[i];
    s += v; mx = fmaxf(mx, v);
  }
  __shared__ float ss[256], sm[256];
  ss[t] = s; sm[t] = mx; __syncthreads();
  for (int st = 128; st > 0; st >>= 1) {
    if (t < st) { ss[t] += ss[t + st]; sm[t] = fmaxf(sm[t], sm[t + st]); }
    __syncthreads();
  }
  if (t == 0) {
    y0[(long)b * 1024 + c2]       = ss[0] * (1.f / NSP);
    y0[(long)b * 1024 + 512 + c2] = sm[0];
  }
}

// y[b,o] = act( sum_i x[b,i]*w[o,i] + bias[o] )   act: 0=gelu, 1=sigmoid
__global__ void cr_fc(const float* __restrict__ x, const float* __restrict__ w,
                      const float* __restrict__ bias, float* __restrict__ y,
                      int In, int Out, int act) {
  int idx = blockIdx.x * blockDim.x + threadIdx.x;
  int b = idx / Out, o = idx - b * Out;
  if (b >= BB) return;
  float s = bias[o];
  const float* xb = x + (long)b * In;
  const float* wr = w + (long)o * In;
  for (int i = 0; i < In; ++i) s += xb[i] * wr[i];
  y[idx] = (act == 0) ? gelu_f(s) : 1.f / (1.f + expf(-s));
}

// fused = rgb*w0 + h*w1 ; xn = LN_channel(fused)
// outputs: fused f32 (b,c,s), xn f32 (b,c,s), xn f16 TRANSPOSED (b,s,c)
__global__ void fuse_ln1(const float* __restrict__ rgb, const float* __restrict__ h,
                         const float* __restrict__ wgt,
                         const float* __restrict__ g, const float* __restrict__ be,
                         float* __restrict__ fused, float* __restrict__ xnf,
                         f16* __restrict__ xnT) {
  int b = blockIdx.x / NSP, s = blockIdx.x - b * NSP;
  int c = threadIdx.x;   // 256
  long idx = ((long)b * CCH + c) * NSP + s;
  float w0 = wgt[(long)b * 512 + c];
  float w1 = wgt[(long)b * 512 + CCH + c];
  float f  = rgb[idx] * w0 + h[idx] * w1;
  fused[idx] = f;
  __shared__ float ss[256], s2[256];
  ss[c] = f; s2[c] = f * f; __syncthreads();
  for (int st = 128; st > 0; st >>= 1) {
    if (c < st) { ss[c] += ss[c + st]; s2[c] += s2[c + st]; }
    __syncthreads();
  }
  float mu  = ss[0] * (1.f / CCH);
  float var = s2[0] * (1.f / CCH) - mu * mu;
  float xv  = (f - mu) * rsqrtf(var + 1e-5f) * g[c] + be[c];
  xnf[idx] = xv;
  xnT[((long)b * NSP + s) * CCH + c] = (f16)xv;   // coalesced
}

// LN over channels -> f16 transposed (b,s,c)
__global__ void ln_chw_f16(const float* __restrict__ x,
                           const float* __restrict__ g, const float* __restrict__ be,
                           f16* __restrict__ outT) {
  int b = blockIdx.x / NSP, s = blockIdx.x - b * NSP;
  int c = threadIdx.x;
  long idx = ((long)b * CCH + c) * NSP + s;
  float v = x[idx];
  __shared__ float ss[256], s2[256];
  ss[c] = v; s2[c] = v * v; __syncthreads();
  for (int st = 128; st > 0; st >>= 1) {
    if (c < st) { ss[c] += ss[c + st]; s2[c] += s2[c + st]; }
    __syncthreads();
  }
  float mu  = ss[0] * (1.f / CCH);
  float var = s2[0] * (1.f / CCH) - mu * mu;
  outT[((long)b * NSP + s) * CCH + c] =
      (f16)((v - mu) * rsqrtf(var + 1e-5f) * g[c] + be[c]);
}

// depthwise 7x7 on q viewed as (B*G, CG, 28, 28), pad 3
__global__ void off_dwconv7(const float* __restrict__ q, const float* __restrict__ w,
                            const float* __restrict__ bias, float* __restrict__ obuf) {
  long idx = (long)blockIdx.x * blockDim.x + threadIdx.x;
  if (idx >= (long)NBH * CG * NSP) return;
  int s  = idx % NSP;
  int cc = (idx / NSP) % CG;
  int bg = idx / ((long)NSP * CG);
  int b = bg >> 3, g = bg & 7;
  int yy = s / WWD, xx = s - yy * WWD;
  const float* qc = q + ((long)b * CCH + g * CG + cc) * NSP;
  const float* wc = w + (long)cc * 49;
  float acc = bias[cc];
  for (int dy = 0; dy < 7; ++dy) {
    int iy = yy + dy - 3; if (iy < 0 || iy >= HH) continue;
    for (int dx = 0; dx < 7; ++dx) {
      int ix = xx + dx - 3; if (ix < 0 || ix >= WWD) continue;
      acc += qc[iy * WWD + ix] * wc[dy * 7 + dx];
    }
  }
  obuf[idx] = acc;
}

// LN(cg) + gelu + 1x1->2 + tanh scale + reference points => pos (y,x)
__global__ void off_ln_pw(const float* __restrict__ obuf,
                          const float* __restrict__ g, const float* __restrict__ be,
                          const float* __restrict__ pw, float* __restrict__ pos) {
  int bg = blockIdx.x / NSP, s = blockIdx.x - bg * NSP;
  int cc = threadIdx.x;  // 32
  float v = obuf[((long)bg * CG + cc) * NSP + s];
  __shared__ float ss[32], s2[32], gv[32];
  ss[cc] = v; s2[cc] = v * v; __syncthreads();
  for (int st = 16; st > 0; st >>= 1) {
    if (cc < st) { ss[cc] += ss[cc + st]; s2[cc] += s2[cc + st]; }
    __syncthreads();
  }
  float mu  = ss[0] * (1.f / CG);
  float var = s2[0] * (1.f / CG) - mu * mu;
  gv[cc] = gelu_f((v - mu) * rsqrtf(var + 1e-5f) * g[cc] + be[cc]);
  __syncthreads();
  if (cc == 0) {
    float oy = 0.f, ox = 0.f;
    for (int i = 0; i < CG; ++i) { oy += gv[i] * pw[i]; ox += gv[i] * pw[CG + i]; }
    oy = tanhf(oy) * (3.f / HH);
    ox = tanhf(ox) * (3.f / WWD);
    int iy = s / WWD, ix = s - iy * WWD;
    float ry = (2 * iy + 1) * (1.f / HH) - 1.f;
    float rx = (2 * ix + 1) * (1.f / WWD) - 1.f;
    pos[((long)bg * NSP + s) * 2 + 0] = oy + ry;
    pos[((long)bg * NSP + s) * 2 + 1] = ox + rx;
  }
}

// xsT[b, s, g*32+cc] = bilinear(xn channel, pos)  (transposed f16)
__global__ void sample_xs(const float* __restrict__ xnf, const float* __restrict__ pos,
                          f16* __restrict__ xsT) {
  long idx = (long)blockIdx.x * blockDim.x + threadIdx.x;
  if (idx >= (long)NBH * CG * NSP) return;
  int s  = idx % NSP;
  int cc = (idx / NSP) % CG;
  int bg = idx / ((long)NSP * CG);
  int b = bg >> 3, g = bg & 7;
  float py = pos[((long)bg * NSP + s) * 2 + 0];
  float px = pos[((long)bg * NSP + s) * 2 + 1];
  float gx = (px + 1.f) * 0.5f * (WWD - 1);
  float gy = (py + 1.f) * 0.5f * (HH - 1);
  const float* img = xnf + ((long)b * CCH + g * CG + cc) * NSP;
  xsT[((long)b * NSP + s) * CCH + g * CG + cc] = (f16)bilin2d(img, WWD, HH, gx, gy);
}

// per (bh, m): add continuous-RPE bias and softmax -> f16 rows of stride KP
__global__ void bias_softmax(const float* __restrict__ attn,
                             const float* __restrict__ pos,
                             const float* __restrict__ rpe,
                             f16* __restrict__ attnp) {
  int bh = blockIdx.x / NSP, m = blockIdx.x - bh * NSP;
  int head = bh & 7;
  int t = threadIdx.x;  // 256
  int my = m / WWD, mx = m - my * WWD;
  float qy = (2 * my + 1) * (1.f / HH) - 1.f;
  float qx = (2 * mx + 1) * (1.f / WWD) - 1.f;
  const float* arow = attn + ((long)bh * NSP + m) * NSP;
  const float* posb = pos + (long)bh * NSP * 2;
  const float* tab  = rpe + (long)head * 55 * 55;
  __shared__ float row[NSP];
  __shared__ float red[256];
  float lmax = -3.402823466e38f;
  for (int nn = t; nn < NSP; nn += 256) {
    float py = posb[nn * 2 + 0], px = posb[nn * 2 + 1];
    float dy = (qy - py) * 0.5f, dx = (qx - px) * 0.5f;
    float gx = (dx + 1.f) * 27.f;   // 0.5*(55-1)
    float gy = (dy + 1.f) * 27.f;
    float v = arow[nn] + bilin2d(tab, 55, 55, gx, gy);
    row[nn] = v;
    lmax = fmaxf(lmax, v);
  }
  red[t] = lmax; __syncthreads();
  for (int st = 128; st > 0; st >>= 1) {
    if (t < st) red[t] = fmaxf(red[t], red[t + st]);
    __syncthreads();
  }
  float mxv = red[0]; __syncthreads();
  float lsum = 0.f;
  for (int nn = t; nn < NSP; nn += 256) {
    float e = expf(row[nn] - mxv);
    row[nn] = e;
    lsum += e;
  }
  red[t] = lsum; __syncthreads();
  for (int st = 128; st > 0; st >>= 1) {
    if (t < st) red[t] += red[t + st];
    __syncthreads();
  }
  float inv = 1.f / red[0];
  f16* orow = attnp + ((long)bh * NSP + m) * KP;
  for (int nn = t; nn < NSP; nn += 256) orow[nn] = (f16)(row[nn] * inv);
  if (t < KP - NSP) orow[NSP + t] = (f16)0;   // zero the K padding
}

// depthwise 3x3 + gelu on (B,1024,28,28) -> f16 transposed (b,s,c4)
__global__ void mlp_dw3(const float* __restrict__ m1, const float* __restrict__ w,
                        const float* __restrict__ bias, f16* __restrict__ m2T) {
  long idx = (long)blockIdx.x * blockDim.x + threadIdx.x;
  if (idx >= (long)BB * 1024 * NSP) return;
  int s = idx % NSP;
  int c = (idx / NSP) % 1024;
  int b = idx / ((long)NSP * 1024);
  int yy = s / WWD, xx = s - yy * WWD;
  const float* src = m1 + ((long)b * 1024 + c) * NSP;
  const float* wc  = w + (long)c * 9;
  float acc = bias[c];
  for (int dy = 0; dy < 3; ++dy) {
    int iy = yy + dy - 1; if (iy < 0 || iy >= HH) continue;
    for (int dx = 0; dx < 3; ++dx) {
      int ix = xx + dx - 1; if (ix < 0 || ix >= WWD) continue;
      acc += src[iy * WWD + ix] * wc[dy * 3 + dx];
    }
  }
  m2T[((long)b * NSP + s) * 1024 + c] = (f16)gelu_f(acc);
}

// ---------------------------------------------------------------- host side
static void gemm(hipStream_t st, int mblk,
                 const f16* A, long ars, long abs1, long abs2,
                 const f16* Bp, long bcs, long bbs1, long bbs2,
                 float* Cf, long crs, long ccs, long cbs1, long cbs2,
                 f16* Ch, long hrs, long hcs, long hbs1, long hbs2,
                 const float* bias, const float* resid, long rbs1, long rbs2,
                 float scale, int M, int N, int K, int zdiv, int batches) {
  int mt  = M >> 4;
  int mtb = mt / mblk;            // exact by construction
  int nt  = N >> 4;
  long waves = (long)mtb * nt * batches;
  int blocks = (int)((waves + 7) / 8);
  if (mblk == 4) {
    wmma_gemm_f16<4><<<blocks, 256, 0, st>>>(
        A, ars, abs1, abs2, Bp, bcs, bbs1, bbs2,
        Cf, crs, ccs, cbs1, cbs2, Ch, hrs, hcs, hbs1, hbs2,
        bias, resid, rbs1, rbs2, scale, K, mtb, nt, zdiv, batches);
  } else {
    wmma_gemm_f16<7><<<blocks, 256, 0, st>>>(
        A, ars, abs1, abs2, Bp, bcs, bbs1, bbs2,
        Cf, crs, ccs, cbs1, cbs2, Ch, hrs, hcs, hbs1, hbs2,
        bias, resid, rbs1, rbs2, scale, K, mtb, nt, zdiv, batches);
  }
}

extern "C" void kernel_launch(void* const* d_in, const int* in_sizes, int n_in,
                              void* d_out, int out_size, void* d_ws, size_t ws_size,
                              hipStream_t stream) {
  (void)in_sizes; (void)n_in; (void)out_size; (void)ws_size;
  const float* rgb      = (const float*)d_in[0];
  const float* hmod     = (const float*)d_in[1];
  const float* cr_w1    = (const float*)d_in[2];
  const float* cr_b1    = (const float*)d_in[3];
  const float* cr_w2    = (const float*)d_in[4];
  const float* cr_b2    = (const float*)d_in[5];
  const float* ln1_g    = (const float*)d_in[6];
  const float* ln1_b    = (const float*)d_in[7];
  const float* ln2_g    = (const float*)d_in[8];
  const float* ln2_b    = (const float*)d_in[9];
  const float* wq       = (const float*)d_in[10];
  const float* bq       = (const float*)d_in[11];
  const float* wk       = (const float*)d_in[12];
  const float* bk       = (const float*)d_in[13];
  const float* wv       = (const float*)d_in[14];
  const float* bv       = (const float*)d_in[15];
  const float* wo       = (const float*)d_in[16];
  const float* bo       = (const float*)d_in[17];
  const float* off_dw_w = (const float*)d_in[18];
  const float* off_dw_b = (const float*)d_in[19];
  const float* off_ln_g = (const float*)d_in[20];
  const float* off_ln_b = (const float*)d_in[21];
  const float* off_pw_w = (const float*)d_in[22];
  const float* rpe      = (const float*)d_in[23];
  const float* mlp_w1   = (const float*)d_in[24];
  const float* mlp_b1   = (const float*)d_in[25];
  const float* mlp_dw_w = (const float*)d_in[26];
  const float* mlp_dw_b = (const float*)d_in[27];
  const float* mlp_w2   = (const float*)d_in[28];
  const float* mlp_b2   = (const float*)d_in[29];

  // ---- workspace layout (256B aligned) ----
  char* ws = (char*)d_ws;
  size_t off = 0;
  auto alloc = [&](size_t bytes) -> char* {
    off = (off + 255) & ~(size_t)255;
    char* p = ws + off;
    off += bytes;
    return p;
  };
  const long ACT = (long)BB * CCH * NSP;        // 802816
  float* y0    = (float*)alloc(4096 * 4);
  float* y1    = (float*)alloc(2048 * 4);
  float* wgt   = (float*)alloc(2048 * 4);
  float* fused = (float*)alloc(ACT * 4);        // (b,c,s)
  float* xnf   = (float*)alloc(ACT * 4);        // (b,c,s)
  f16*   xnT   = (f16*)  alloc(ACT * 2);        // (b,s,c)
  float* qf    = (float*)alloc(ACT * 4);        // (b,c,s)
  f16*   qT    = (f16*)  alloc(ACT * 2);        // (b,s,c)
  float* obuf  = (float*)alloc(ACT * 4);
  float* pos   = (float*)alloc((long)NBH * NSP * 2 * 4);
  f16*   xsT   = (f16*)  alloc(ACT * 2);        // (b,s,c)
  f16*   kT    = (f16*)  alloc(ACT * 2);        // (b,s,c)
  f16*   vpad  = (f16*)  alloc((long)BB * CCH * KP * 2);   // (b,c,KP)
  f16*   outT  = (f16*)  alloc(ACT * 2);        // (b,s,c)
  f16*   wq16  = (f16*)  alloc(65536 * 2);
  f16*   wk16  = (f16*)  alloc(65536 * 2);
  f16*   wv16  = (f16*)  alloc(65536 * 2);
  f16*   wo16  = (f16*)  alloc(65536 * 2);
  f16*   w116  = (f16*)  alloc(262144 * 2);
  f16*   w216  = (f16*)  alloc(262144 * 2);
  const long ATT = (long)NBH * NSP * NSP;       // 19,668,992
  float* attnf  = (float*)alloc(ATT * 4);       // logits (dead after softmax)
  f16*   attnpp = (f16*)  alloc((long)NBH * NSP * KP * 2); // padded probs
  float* x1f    = (float*)alloc(ACT * 4);       // (b,c,s)
  f16*   xn2T   = (f16*)  alloc(ACT * 2);       // (b,s,c)
  // MLP intermediates alias the (dead-by-then) logits region
  const long MLPN = (long)BB * 1024 * NSP;      // 3,211,264
  float* m1f = attnf;                           // (b,c4,s)  12.85 MB
  f16*   m2T = (f16*)((char*)attnf + ((MLPN * 4 + 255) & ~(size_t)255)); // (b,s,c4)

  // ---- weight conversion ----
  cvt_f32_to_f16<<<(65536 + 255) / 256, 256, 0, stream>>>(wq, wq16, 65536);
  cvt_f32_to_f16<<<(65536 + 255) / 256, 256, 0, stream>>>(wk, wk16, 65536);
  cvt_f32_to_f16<<<(65536 + 255) / 256, 256, 0, stream>>>(wv, wv16, 65536);
  cvt_f32_to_f16<<<(65536 + 255) / 256, 256, 0, stream>>>(wo, wo16, 65536);
  cvt_f32_to_f16<<<(262144 + 255) / 256, 256, 0, stream>>>(mlp_w1, w116, 262144);
  cvt_f32_to_f16<<<(262144 + 255) / 256, 256, 0, stream>>>(mlp_w2, w216, 262144);
  zero_pad_v<<<(BB * CCH * 16 + 255) / 256, 256, 0, stream>>>(vpad);

  // ---- channel rectify ----
  cr_pool<<<BB * 512, 256, 0, stream>>>(rgb, hmod, y0);
  cr_fc<<<(BB * 512 + 255) / 256, 256, 0, stream>>>(y0, cr_w1, cr_b1, y1, 1024, 512, 0);
  cr_fc<<<(BB * 512 + 255) / 256, 256, 0, stream>>>(y1, cr_w2, cr_b2, wgt, 512, 512, 1);

  // ---- fuse + LN1 ----
  fuse_ln1<<<BB * NSP, 256, 0, stream>>>(rgb, hmod, wgt, ln1_g, ln1_b, fused, xnf, xnT);

  // ---- q = wq @ xn + bq : Cf=(b,c,s), Ch=(b,s,c) ----
  gemm(stream, 4, wq16, 256, 0, 0, xnT, 256, 200704, 0,
       qf, 784, 1, 200704, 0, qT, 1, 256, 200704, 0,
       bq, nullptr, 0, 0, 1.f, 256, 784, 256, 1, BB);

  // ---- offsets ----
  off_dwconv7<<<(int)(((long)NBH * CG * NSP + 255) / 256), 256, 0, stream>>>(
      qf, off_dw_w, off_dw_b, obuf);
  off_ln_pw<<<NBH * NSP, 32, 0, stream>>>(obuf, off_ln_g, off_ln_b, off_pw_w, pos);

  // ---- deformable sampling -> (b,s,c) f16 ----
  sample_xs<<<(int)(((long)NBH * CG * NSP + 255) / 256), 256, 0, stream>>>(xnf, pos, xsT);

  // ---- k -> (b,s,c) ; v -> (b,c,KP) ----
  gemm(stream, 4, wk16, 256, 0, 0, xsT, 256, 200704, 0,
       nullptr, 0, 0, 0, 0, kT, 1, 256, 200704, 0,
       bk, nullptr, 0, 0, 1.f, 256, 784, 256, 1, BB);
  gemm(stream, 4, wv16, 256, 0, 0, xsT, 256, 200704, 0,
       nullptr, 0, 0, 0, 0, vpad, KP, 1, (long)CCH * KP, 0,
       bv, nullptr, 0, 0, 1.f, 256, 784, 256, 1, BB);

  // ---- attention logits: z=(b,head), A=qT rows (k=hc contiguous), B=kT ----
  gemm(stream, 7, qT, 256, 200704, 32, kT, 256, 200704, 32,
       attnf, 784, 1, 8L * 614656, 614656, nullptr, 0, 0, 0, 0,
       nullptr, nullptr, 0, 0, 0.176776695296636893f, 784, 784, 32, 8, NBH);

  // ---- bias + softmax -> padded f16 probs ----
  bias_softmax<<<NBH * NSP, 256, 0, stream>>>(attnf, pos, rpe, attnpp);

  // ---- out = attn @ v : K padded to 800, D -> outT (b,s,c) ----
  gemm(stream, 7, attnpp, KP, 8L * NSP * KP, (long)NSP * KP,
       vpad, KP, (long)CCH * KP, 32L * KP,
       nullptr, 0, 0, 0, 0, outT, 256, 1, 200704, 32,
       nullptr, nullptr, 0, 0, 1.f, 784, 32, KP, 8, NBH);

  // ---- x1 = wo @ out + bo + fused ----
  gemm(stream, 4, wo16, 256, 0, 0, outT, 256, 200704, 0,
       x1f, 784, 1, 200704, 0, nullptr, 0, 0, 0, 0,
       bo, fused, 200704, 0, 1.f, 256, 784, 256, 1, BB);

  // ---- ConvMLP ----
  ln_chw_f16<<<BB * NSP, 256, 0, stream>>>(x1f, ln2_g, ln2_b, xn2T);
  gemm(stream, 4, w116, 256, 0, 0, xn2T, 256, 200704, 0,
       m1f, 784, 1, 802816, 0, nullptr, 0, 0, 0, 0,
       mlp_b1, nullptr, 0, 0, 1.f, 1024, 784, 256, 1, BB);
  mlp_dw3<<<(int)((MLPN + 255) / 256), 256, 0, stream>>>(m1f, mlp_dw_w, mlp_dw_b, m2T);
  gemm(stream, 4, w216, 1024, 0, 0, m2T, 1024, 802816, 0,
       (float*)d_out, 784, 1, 200704, 0, nullptr, 0, 0, 0, 0,
       mlp_b2, x1f, 200704, 0, 1.f, 256, 784, 1024, 1, BB);
}
